// DynamicWeights_49039936586101
// MI455X (gfx1250) — matrix operand order repaired and checked by reading, not compiled, not run on hardware
//
#include <hip/hip_runtime.h>

typedef __attribute__((ext_vector_type(16))) _Float16 v16h;
typedef __attribute__((ext_vector_type(8)))  float    v8f;

constexpr int kN = 2, kH = 128, kW = 128, kP = kH * kW;

#define WMMA(a, b, c) \
  __builtin_amdgcn_wmma_f32_16x16x32_f16(false, (a), false, (b), (short)0, (c), false, false)

// Pack two f32 -> adjacent f16 pair, single 32-bit store
__device__ __forceinline__ void pack2(_Float16* d, float f0, float f1) {
  union { _Float16 h[2]; unsigned u; } pk;
  pk.h[0] = (_Float16)f0; pk.h[1] = (_Float16)f1;
  *(unsigned*)d = pk.u;
}

// ---------------- weight prep: f32 MxK row-major -> f16 fragment-tiled ----------------
// Tile (mt,kt) of 16x32; per lane 16 contiguous halves in ISA 7.12.2 A-layout:
//   m = mt*16 + (lane&15);  k(e) = kt*32 + (e>>3)*16 + 8*(lane>=16) + (e&7)
__global__ __launch_bounds__(256) void k_prep(const float* __restrict__ W,
                                              _Float16* __restrict__ D, int M, int K) {
  const int total = (M >> 4) * (K >> 5) * 32;
  const int idx = blockIdx.x * 256 + threadIdx.x;
  if (idx >= total) return;
  const int lane = idx & 31, tile = idx >> 5;
  const int numKt = K >> 5;
  const int mt = tile / numKt, kt = tile - mt * numKt;
  const int m = mt * 16 + (lane & 15);
  const int kh = (lane >> 4) << 3;
  v16h v;
#pragma unroll
  for (int e = 0; e < 16; ++e)
    v[e] = (_Float16)W[m * K + kt * 32 + ((e >> 3) << 4) + kh + (e & 7)];
  *(v16h*)(D + (size_t)idx * 16) = v;
}

// fragment-order LDS address for B element (row rr in 0..31 within k-step, col nn)
// region layout: [(ntile)*32 + lane]*16 + e ; lane = ((rr&8)<<1)+(nn&15), e = (rr&7)+((rr&16)>>1)
__device__ __forceinline__ int b_addr(int ntile, int rr, int nn) {
  return ((ntile * 32) + ((rr & 8) << 1) + (nn & 15)) * 16 + (rr & 7) + ((rr & 16) >> 1);
}

// ---------------- generic 1x1-conv GEMM: O[n,row,p] = A[row,:] . B[n,:,p] ----------------
// Ap: fragment-prepped f16 weights. mode: 0 none, 1 bn+relu(p0=g,p1=b), 2 bias+relu(p0)
__global__ __launch_bounds__(256) void k_gemm(
    const _Float16* __restrict__ Ap, const float* __restrict__ B, int K,
    long bStrideB, float* __restrict__ O, long bStrideO,
    const float* __restrict__ p0, const float* __restrict__ p1, int mode)
{
  __shared__ __attribute__((aligned(32))) _Float16 Bsh[2 * 32 * 16];  // 2 n-tiles
  const int tid = threadIdx.x, lane = tid & 31, wv = tid >> 5;
  const int mt = wv >> 1, nt = wv & 1;
  const int pBase = blockIdx.x * 32;
  const int mBlock = blockIdx.y * 64;
  const int n = blockIdx.z;
  const int numKt = K >> 5;
  v8f acc = {};
  for (int kt = 0; kt < numKt; ++kt) {
    // stage 32K x 32px panel, pair-packed into fragment order
    for (int t = tid; t < 512; t += 256) {
      int kp = t >> 5, nn = t & 31;           // rows 2kp, 2kp+1
      int rr = kp * 2;
      const float* bp = &B[(long)n * bStrideB + (long)(kt * 32 + rr) * kP + pBase + nn];
      pack2(&Bsh[b_addr(nn >> 4, rr, nn)], bp[0], bp[kP]);
    }
    if (kt + 1 < numKt)
      __builtin_prefetch(&B[(long)n * bStrideB + (long)((kt + 1) * 32 + (tid >> 3)) * kP +
                            pBase + ((tid & 7) << 2)], 0, 0);
    __syncthreads();
    v16h a = *(const v16h*)(Ap + (((size_t)((mBlock / 16 + mt) * numKt + kt)) * 32 + lane) * 16);
    v16h b = *(const v16h*)(Bsh + ((size_t)(nt * 32 + lane)) * 16);
    acc = WMMA(a, b, acc);
    __syncthreads();
  }
  const int col = pBase + nt * 16 + (lane & 15);
  const int rB = mBlock + mt * 16 + ((lane >> 4) << 3);
#pragma unroll
  for (int r = 0; r < 8; ++r) {
    int row = rB + r;
    float v = acc[r];
    if (mode == 1)      v = fmaxf(v * (p0[row] * rsqrtf(1.f + 1e-5f)) + p1[row], 0.f);
    else if (mode == 2) v = fmaxf(v + p0[row], 0.f);
    O[(long)n * bStrideO + (long)row * kP + col] = v;
  }
}

// ---------------- deformable conv as implicit GEMM (M=64, K=576=c*9+k, N=32 px) ----------------
__global__ __launch_bounds__(256) void k_dcn(
    const float* __restrict__ src, long srcBS,
    const float* __restrict__ off, long offBS,
    const _Float16* __restrict__ Ap,          // prepped 64x576 weights (4 x 18 tiles)
    const float* __restrict__ bias,
    const float* __restrict__ resid, long residBS,
    float* __restrict__ out, long outBS)
{
  __shared__ __attribute__((aligned(32))) _Float16 Bsh[18 * 2 * 32 * 16];  // 36 KB
  __shared__ int   pi[9 * 32 * 2];
  __shared__ float pw[9 * 32 * 2];
  const int tid = threadIdx.x, lane = tid & 31, wv = tid >> 5;
  const int pBase = blockIdx.x * 32, n = blockIdx.z;

  // per-(tap,pixel) bilinear coordinates
  for (int t = tid; t < 288; t += 256) {
    int k = t >> 5, px = t & 31;
    int p = pBase + px, h = p >> 7, w = p & 127;
    float oy = off[(long)n * offBS + (long)(2 * k) * kP + p];
    float ox = off[(long)n * offBS + (long)(2 * k + 1) * kP + p];
    float yy = oy + (float)(h + (k / 3) - 1);
    float xx = ox + (float)(w + (k % 3) - 1);
    float y0 = floorf(yy), x0 = floorf(xx);
    pi[2 * t] = (int)y0;  pi[2 * t + 1] = (int)x0;
    pw[2 * t] = yy - y0;  pw[2 * t + 1] = xx - x0;
  }
  __syncthreads();

  // build B panel in fragment order; two K-rows per thread-iteration, packed store
  for (int t = tid; t < 288 * 32; t += 256) {
    int rp = t >> 5, px = t & 31;             // rows r0=2rp, r0+1
    int r0 = rp * 2;
    float v2[2];
#pragma unroll
    for (int j = 0; j < 2; ++j) {
      int r = r0 + j;
      int c = r / 9, k = r - 9 * c;
      int q = (k << 5) + px;
      int y0 = pi[2 * q], x0 = pi[2 * q + 1];
      float wy = pw[2 * q], wx = pw[2 * q + 1];
      const float* img = src + (long)n * srcBS + (long)c * kP;
      bool yv0 = (y0 >= 0) & (y0 < kH), yv1 = (y0 + 1 >= 0) & (y0 + 1 < kH);
      bool xv0 = (x0 >= 0) & (x0 < kW), xv1 = (x0 + 1 >= 0) & (x0 + 1 < kW);
      float v00 = (yv0 & xv0) ? img[y0 * kW + x0] : 0.f;
      float v01 = (yv0 & xv1) ? img[y0 * kW + x0 + 1] : 0.f;
      float v10 = (yv1 & xv0) ? img[(y0 + 1) * kW + x0] : 0.f;
      float v11 = (yv1 & xv1) ? img[(y0 + 1) * kW + x0 + 1] : 0.f;
      v2[j] = v00 * (1.f - wy) * (1.f - wx) + v01 * (1.f - wy) * wx
            + v10 * wy * (1.f - wx) + v11 * wy * wx;
    }
    int kt = r0 >> 5, rr = r0 & 31;
    pack2(&Bsh[(kt * 2) * 512 + b_addr(px >> 4, rr, px)], v2[0], v2[1]);
  }
  __syncthreads();

  const int mt = wv >> 1, nt = wv & 1;
  v8f acc = {};
#pragma unroll 3
  for (int ks = 0; ks < 18; ++ks) {
    v16h a = *(const v16h*)(Ap + (((size_t)(mt * 18 + ks)) * 32 + lane) * 16);
    v16h b = *(const v16h*)(Bsh + ((size_t)((ks * 2 + nt) * 32 + lane)) * 16);
    acc = WMMA(a, b, acc);
  }
  const int col = pBase + nt * 16 + (lane & 15);
  const int rB = mt * 16 + ((lane >> 4) << 3);
#pragma unroll
  for (int r = 0; r < 8; ++r) {
    int row = rB + r;
    float v = acc[r];
    if (bias)  v += bias[row];
    if (resid) v += resid[(long)n * residBS + (long)row * kP + col];
    out[(long)n * outBS + (long)row * kP + col] = v;
  }
}

// ---------------- direct 3x3 conv, small Co, register-blocked, weights in LDS ----------------
template <int CO>
__global__ __launch_bounds__(256) void k_conv3(
    const float* __restrict__ in, long inBS,
    const float* __restrict__ Wt,
    const float* __restrict__ bnG, const float* __restrict__ bnB,
    float* __restrict__ out, long outBS)
{
  extern __shared__ float Wsh[];
  const int tid = threadIdx.x;
  for (int t = tid; t < CO * 576; t += 256) Wsh[t] = Wt[t];
  __syncthreads();
  const int idx = blockIdx.x * 256 + tid;
  const int p = idx & (kP - 1);
  const int n = idx >> 14;
  const int h = p >> 7, w = p & 127;
  float acc[CO];
#pragma unroll
  for (int co = 0; co < CO; ++co) acc[co] = 0.f;
  for (int ci = 0; ci < 64; ++ci) {
    const float* img = in + (long)n * inBS + (long)ci * kP;
    float v[9];
#pragma unroll
    for (int t9 = 0; t9 < 9; ++t9) {
      int y = h + t9 / 3 - 1, x = w + t9 % 3 - 1;
      v[t9] = (y >= 0 && y < kH && x >= 0 && x < kW) ? img[y * kW + x] : 0.f;
    }
#pragma unroll
    for (int co = 0; co < CO; ++co) {
      const float* wr = Wsh + (co * 64 + ci) * 9;
      float s = 0.f;
#pragma unroll
      for (int t9 = 0; t9 < 9; ++t9) s = fmaf(v[t9], wr[t9], s);
      acc[co] += s;
    }
  }
#pragma unroll
  for (int co = 0; co < CO; ++co) {
    float val = acc[co];
    if (bnG) val = fmaxf(val * (bnG[co] * rsqrtf(1.f + 1e-5f)) + bnB[co], 0.f);
    out[(long)n * outBS + (long)co * kP + p] = val;
  }
}

// ---------------- wkp[n,k,p] = softmax_k(aff)[k] * sigmoid(of[18+k]) ----------------
__global__ __launch_bounds__(256) void k_wkp(
    const float* __restrict__ aff, const float* __restrict__ ofb, float* __restrict__ wkp)
{
  const int idx = blockIdx.x * 256 + threadIdx.x;
  const int p = idx & (kP - 1);
  const int n = idx >> 14;
  float a[9], m = -1e30f;
#pragma unroll
  for (int k = 0; k < 9; ++k) {
    a[k] = aff[(long)n * 9 * kP + (long)k * kP + p];
    m = fmaxf(m, a[k]);
  }
  float s = 0.f;
#pragma unroll
  for (int k = 0; k < 9; ++k) { a[k] = __expf(a[k] - m); s += a[k]; }
  float inv = 1.f / s;
#pragma unroll
  for (int k = 0; k < 9; ++k) {
    float f = ofb[(long)n * 27 * kP + (long)(18 + k) * kP + p];
    float sg = 1.f / (1.f + __expf(-f));
    wkp[(long)n * 9 * kP + (long)k * kP + p] = a[k] * inv * sg;
  }
}

// ------- fused upsp GEMM (576x64 x 64x16) + weighted-K reduction -> cat channels -------
__global__ __launch_bounds__(128) void k_upsp(
    const float* __restrict__ xu, const _Float16* __restrict__ Ap,  // prepped 576x64 (36x2 tiles)
    const float* __restrict__ Ub, const float* __restrict__ wkp,
    float* __restrict__ cat, int coBase)
{
  __shared__ __attribute__((aligned(32))) _Float16 Bsh[2 * 32 * 16];  // 2 k-steps, 16 px
  __shared__ float Ysh[4 * 144 * 16];
  const int tid = threadIdx.x, lane = tid & 31, wv = tid >> 5;
  const int pBase = blockIdx.x * 16, n = blockIdx.z;
  // stage xu (64K x 16px) in fragment order, pair-packed
  for (int t = tid; t < 512; t += 128) {
    int kp = t >> 4, px = t & 15;             // rows 2kp, 2kp+1
    int r0 = kp * 2;
    const float* bp = &xu[(long)n * 64 * kP + (long)r0 * kP + pBase + px];
    int kt = r0 >> 5, rr = r0 & 31;
    pack2(&Bsh[kt * 512 + b_addr(0, rr, px)], bp[0], bp[kP]);
  }
  __syncthreads();
  v16h b0 = *(const v16h*)(Bsh + (size_t)lane * 16);
  v16h b1 = *(const v16h*)(Bsh + (size_t)(32 + lane) * 16);
  v8f acc[9];
#pragma unroll
  for (int mt = 0; mt < 9; ++mt) { v8f z = {}; acc[mt] = z; }
#pragma unroll
  for (int mt = 0; mt < 9; ++mt) {
    v16h a0 = *(const v16h*)(Ap + (((size_t)((wv * 9 + mt) * 2 + 0)) * 32 + lane) * 16);
    acc[mt] = WMMA(a0, b0, acc[mt]);
    v16h a1 = *(const v16h*)(Ap + (((size_t)((wv * 9 + mt) * 2 + 1)) * 32 + lane) * 16);
    acc[mt] = WMMA(a1, b1, acc[mt]);
  }
  float* Y = Ysh + wv * 144 * 16;  // wave owns rows [wv*144, wv*144+144)
#pragma unroll
  for (int mt = 0; mt < 9; ++mt)
#pragma unroll
    for (int r = 0; r < 8; ++r)
      Y[(mt * 16 + r + ((lane >> 4) << 3)) * 16 + (lane & 15)] = acc[mt][r];
  __syncthreads();
  for (int t = lane; t < 256; t += 32) {
    int cl = t >> 4, px = t & 15;
    int c = wv * 16 + cl;
    float s = 0.f;
#pragma unroll
    for (int k = 0; k < 9; ++k)
      s += (Y[(cl * 9 + k) * 16 + px] + Ub[c * 9 + k]) *
           wkp[(long)n * 9 * kP + (long)k * kP + pBase + px];
    cat[(long)n * 256 * kP + (long)(coBase + c) * kP + pBase + px] = s;
  }
}

__global__ void k_copy4(const float4* __restrict__ src, float4* __restrict__ dst, int n4) {
  int i = blockIdx.x * 256 + threadIdx.x;
  if (i < n4) dst[i] = src[i];
}

extern "C" void kernel_launch(void* const* d_in, const int* in_sizes, int n_in,
                              void* d_out, int out_size, void* d_ws, size_t ws_size,
                              hipStream_t stream) {
  (void)in_sizes; (void)n_in; (void)out_size; (void)ws_size;
  const float* rgb = (const float*)d_in[0];
  const float* depth[3] = {(const float*)d_in[1], (const float*)d_in[2], (const float*)d_in[3]};
  const float* smooth_w   = (const float*)d_in[4];
  const float* smooth_g   = (const float*)d_in[5];
  const float* smooth_b   = (const float*)d_in[6];
  const float* warp_off_w = (const float*)d_in[7];
  const float* warp_dcn_w = (const float*)d_in[8];
  const float* warp_c1_w  = (const float*)d_in[9];
  const float* warp_bn_g  = (const float*)d_in[10];
  const float* warp_bn_b  = (const float*)d_in[11];
  const float* cat_off_w  = (const float*)d_in[12];
  const float* unfold_w   = (const float*)d_in[13];
  const float* unfold_b   = (const float*)d_in[14];
  const float* upsp_w     = (const float*)d_in[15];
  const float* upsp_b     = (const float*)d_in[16];
  const float* scale2_w   = (const float*)d_in[17];
  const float* scale2_b   = (const float*)d_in[18];
  float* outp = (float*)d_out;

  float* ws = (float*)d_ws;
  size_t o = 0;
  float* cat   = ws + o; o += (size_t)kN * 256 * kP;  // [x | out1 | out2 | out3]
  float* ofb   = ws + o; o += (size_t)kN * 27 * kP;
  float* d2    = ws + o; o += (size_t)kN * 64 * kP;
  float* aff   = ws + o; o += (size_t)kN * 9 * kP;
  float* off18 = ws + o; o += (size_t)kN * 18 * kP;
  float* wkp   = ws + o; o += (size_t)kN * 9 * kP;
  float* xu    = ws + o; o += (size_t)kN * 64 * kP;

  // fragment-prepped f16 weights (after float region; kP-multiple offset -> 32B aligned)
  _Float16* hb = (_Float16*)(ws + o);
  size_t ho = 0;
  _Float16* smoothP = hb + ho; ho += (size_t)4 * 8 * 512;    // 64x256
  _Float16* scale2P = hb + ho; ho += (size_t)16 * 8 * 512;   // 256x256
  _Float16* warpP[3], *unfoldP[3], *upspP[3];
  for (int i = 0; i < 3; ++i) {
    warpP[i]   = hb + ho; ho += (size_t)4 * 18 * 512;        // 64x576
    unfoldP[i] = hb + ho; ho += (size_t)4 * 18 * 512;        // 64x576
    upspP[i]   = hb + ho; ho += (size_t)36 * 2 * 512;        // 576x64
  }

  auto prep = [&](const float* W, _Float16* D, int M, int K) {
    int threads = (M >> 4) * (K >> 5) * 32;
    k_prep<<<(threads + 255) / 256, 256, 0, stream>>>(W, D, M, K);
  };
  prep(smooth_w, smoothP, 64, 256);
  prep(scale2_w, scale2P, 256, 256);
  for (int i = 0; i < 3; ++i) {
    prep(warp_dcn_w + (size_t)i * 64 * 576, warpP[i], 64, 576);
    prep(unfold_w + (size_t)i * 64 * 576, unfoldP[i], 64, 576);
    prep(upsp_w + (size_t)i * 576 * 64, upspP[i], 576, 64);
  }

  // x = relu(bn(smooth_w @ rgb))  -> cat channels 0..63
  k_gemm<<<dim3(kP / 32, 1, kN), 256, 0, stream>>>(
      smoothP, rgb, 256, (long)256 * kP, cat, (long)256 * kP, smooth_g, smooth_b, 1);

  for (int i = 0; i < 3; ++i) {
    k_conv3<27><<<dim3((kN * kP) / 256), 256, 27 * 576 * sizeof(float), stream>>>(
        depth[i], (long)64 * kP, warp_off_w + (size_t)i * 27 * 576, nullptr, nullptr,
        ofb, (long)27 * kP);
    k_dcn<<<dim3(kP / 32, 1, kN), 256, 0, stream>>>(
        depth[i], (long)64 * kP, ofb, (long)27 * kP, warpP[i],
        nullptr, depth[i], (long)64 * kP, d2, (long)64 * kP);
    k_conv3<9><<<dim3((kN * kP) / 256), 256, 9 * 576 * sizeof(float), stream>>>(
        d2, (long)64 * kP, warp_c1_w + (size_t)i * 9 * 576,
        warp_bn_g + i * 9, warp_bn_b + i * 9, aff, (long)9 * kP);
    k_conv3<18><<<dim3((kN * kP) / 256), 256, 18 * 576 * sizeof(float), stream>>>(
        cat, (long)256 * kP, cat_off_w + (size_t)i * 18 * 576, nullptr, nullptr,
        off18, (long)18 * kP);
    k_wkp<<<dim3((kN * kP) / 256), 256, 0, stream>>>(aff, ofb, wkp);
    k_dcn<<<dim3(kP / 32, 1, kN), 256, 0, stream>>>(
        cat, (long)256 * kP, off18, (long)18 * kP, unfoldP[i],
        unfold_b + i * 64, nullptr, 0, xu, (long)64 * kP);
    k_upsp<<<dim3(kP / 16, 1, kN), 128, 0, stream>>>(
        xu, upspP[i], upsp_b + i * 576, wkp, cat, 64 * (i + 1));
  }

  // s = relu(scale2_w @ cat + b)  -> first half of output
  k_gemm<<<dim3(kP / 32, 4, kN), 256, 0, stream>>>(
      scale2P, cat, 256, (long)256 * kP, outp, (long)256 * kP, scale2_b, nullptr, 2);

  // second half of output = rgb_feat
  int n4 = (kN * 256 * kP) / 4;
  k_copy4<<<(n4 + 255) / 256, 256, 0, stream>>>(
      (const float4*)rgb, (float4*)(outp + (size_t)2 * 256 * kP), n4);
}
